// BiLstmCellEncoder_9792525435317
// MI455X (gfx1250) — matrix operand order repaired and checked by reading, not compiled, not run on hardware
//
#include <hip/hip_runtime.h>
#include <hip/hip_bf16.h>
#include <cstdint>

typedef __attribute__((ext_vector_type(16))) __bf16 v16bf;
typedef __attribute__((ext_vector_type(8)))  __bf16 v8bf;
typedef __attribute__((ext_vector_type(8)))  float  v8f;

#define B_   64
#define S_   512
#define D_   256
#define H_   512
#define G_   2048          // 4H gate rows
#define KTOT 768           // H + D  (concatenated K)
#define NWG  32            // 16 slices x 2 directions
#define UPW  32            // hidden units per WG
#define LROWS 128          // gate rows per WG (4*UPW)
#define KP   784           // LDS K stride (bf16), 32B-aligned rows, bank spread
#define GP   129           // LDS gates stride (f32), conflict-free pointwise reads

static const int LDS_BYTES = LROWS * KP * 2 + B_ * GP * 4;   // 200704 + 33024 = 233728

// ---------------- packing kernels ----------------

// Pack [Whh_row(512) | Wih_row(256)] -> bf16, rows reordered so each WG's
// 128 rows are (gate g, unit jj) = local (g*32+jj) for its unit slice.
__global__ void pack_weights(const float* __restrict__ fWih, const float* __restrict__ fWhh,
                             const float* __restrict__ fbih, const float* __restrict__ fbhh,
                             const float* __restrict__ bWih, const float* __restrict__ bWhh,
                             const float* __restrict__ bbih, const float* __restrict__ bbhh,
                             __bf16* __restrict__ WtP, float* __restrict__ biasP)
{
    const int p   = blockIdx.x & (G_ - 1);
    const int dir = blockIdx.x >> 11;
    const int slice = p >> 7, local = p & 127;
    const int g = local >> 5, jj = local & 31;
    const int orig = g * H_ + slice * UPW + jj;
    const float* Whh = dir ? bWhh : fWhh;
    const float* Wih = dir ? bWih : fWih;
    __bf16* dst = WtP + ((size_t)dir * G_ + p) * KTOT;
    for (int k = threadIdx.x; k < KTOT; k += 256) {
        float v = (k < H_) ? Whh[(size_t)orig * H_ + k]
                           : Wih[(size_t)orig * D_ + (k - H_)];
        dst[k] = (__bf16)v;
    }
    if (threadIdx.x == 0) {
        const float* bi = dir ? bbih : fbih;
        const float* bh = dir ? bbhh : fbhh;
        biasP[dir * G_ + p] = bi[orig] + bh[orig];
    }
}

// x[b][t][d] (f32) -> xbf[t][b][d] (bf16): per-timestep contiguous A operand
__global__ void pack_x(const float* __restrict__ x, __bf16* __restrict__ xbf)
{
    size_t i = (size_t)blockIdx.x * 256 + threadIdx.x;
    if (i >= (size_t)S_ * B_ * D_) return;
    int d = (int)(i & (D_ - 1));
    int b = (int)((i >> 8) & (B_ - 1));
    int t = (int)(i >> 14);
    xbf[((size_t)t * B_ + b) * D_ + d] = (__bf16)x[((size_t)b * S_ + t) * D_ + d];
}

__global__ void pack_attw(const float* __restrict__ w, __bf16* __restrict__ o)
{
    int i = blockIdx.x * 256 + threadIdx.x;     // 64*1024
    o[i] = (__bf16)w[i];
}

// ---------------- persistent bi-LSTM kernel ----------------

__global__ void __launch_bounds__(256) lstm_persistent(
    const __bf16* __restrict__ WtP, const float* __restrict__ biasP,
    const __bf16* __restrict__ xbf, __bf16* __restrict__ hbuf,
    __bf16* __restrict__ hist, unsigned* __restrict__ ctr)
{
    extern __shared__ char smem[];
    __bf16* wlds = (__bf16*)smem;                         // [128][KP] bf16
    float*  gates = (float*)(smem + LROWS * KP * 2);      // [64][GP] f32

    const int dir   = blockIdx.x >> 4;
    const int slice = blockIdx.x & 15;
    const int tid   = threadIdx.x;

    // Stage this WG's packed weight slice [128 x 768] into LDS (row stride KP)
    // via the gfx1250 async-to-LDS DMA path (no VGPR round trip; ASYNCcnt).
    {
        const char* wsrc = (const char*)(WtP + ((size_t)dir * G_ + (size_t)slice * LROWS) * KTOT);
        const unsigned lds_base = (unsigned)(size_t)(void*)wlds;
        // 128 rows x 1536 bytes = 12288 chunks of 16B; all 256 lanes active
        for (int idx = tid; idx < LROWS * 96; idx += 256) {
            int row = idx / 96, col = idx - row * 96;          // col in 16B units
            unsigned ldsoff = lds_base + (unsigned)(row * (KP * 2) + col * 16);
            uint64_t gaddr  = (uint64_t)(size_t)(wsrc + (size_t)row * (KTOT * 2) + (size_t)col * 16);
            asm volatile("global_load_async_to_lds_b128 %0, %1, off"
                         :: "v"(ldsoff), "v"(gaddr) : "memory");
        }
        asm volatile("s_wait_asynccnt 0" ::: "memory");
    }
    __syncthreads();

    const int lane = tid & 31;
    const int wv   = tid >> 5;
    const int mt   = wv & 3;        // M tile (batch)
    const int nb   = wv >> 2;       // N half
    const int l15  = lane & 15;
    const int lh   = lane >> 4;
    const int arow = mt * 16 + l15; // batch row this lane loads

    const int pb  = tid & 63;            // pointwise: batch index
    const int pj0 = (tid >> 6) * 8;      // pointwise: unit base
    const float* biasW = biasP + dir * G_ + slice * LROWS;

    float c[8];
#pragma unroll
    for (int m = 0; m < 8; ++m) c[m] = 0.f;

    for (int t = 0; t < S_; ++t) {
        const int tx = dir ? (S_ - 1 - t) : t;
        const __bf16* hprev = hbuf + (size_t)((t & 1) * 2 + dir) * (B_ * H_);
        __bf16*       hnext = hbuf + (size_t)((((t + 1) & 1)) * 2 + dir) * (B_ * H_);
        const __bf16* hrow = hprev + (size_t)arow * H_;
        const __bf16* xrow = xbf + ((size_t)tx * B_ + arow) * D_;

        v8f zero = {};
        v8f acc[4];
#pragma unroll
        for (int j = 0; j < 4; ++j) acc[j] = zero;

        // gates[64 x 128] = [h | x_t](64x768) @ W_slice^T(768x128)
        for (int kt = 0; kt < 24; ++kt) {
            const __bf16* rb = (kt < 16) ? hrow : (xrow - H_);
            v8bf a0 = *(const v8bf*)(rb + kt * 32 + 16 * lh);
            v8bf a1 = *(const v8bf*)(rb + kt * 32 + 16 * lh + 8);
            v16bf a = __builtin_shufflevector(a0, a1, 0,1,2,3,4,5,6,7,8,9,10,11,12,13,14,15);
#pragma unroll
            for (int j = 0; j < 4; ++j) {
                const __bf16* bp = wlds + ((nb * 4 + j) * 16 + l15) * KP + kt * 32 + 16 * lh;
                v16bf bt = *(const v16bf*)bp;
                acc[j] = __builtin_amdgcn_wmma_f32_16x16x32_bf16(
                    false, a, false, bt, (short)0, acc[j], false, false);
            }
        }

        // D layout: VGPR r -> row mt*16 + r + 8*lh, col (nb*4+j)*16 + l15
#pragma unroll
        for (int j = 0; j < 4; ++j)
#pragma unroll
            for (int r = 0; r < 8; ++r)
                gates[(mt * 16 + r + 8 * lh) * GP + (nb * 4 + j) * 16 + l15] = acc[j][r];
        __syncthreads();

        // pointwise LSTM cell update; c lives in registers
#pragma unroll
        for (int m = 0; m < 8; ++m) {
            int jj = pj0 + m;
            float gi = gates[pb * GP + jj]           + biasW[jj];
            float gf = gates[pb * GP + UPW + jj]     + biasW[UPW + jj];
            float gg = gates[pb * GP + 2 * UPW + jj] + biasW[2 * UPW + jj];
            float go = gates[pb * GP + 3 * UPW + jj] + biasW[3 * UPW + jj];
            float ig = 1.f / (1.f + __expf(-gi));
            float fg = 1.f / (1.f + __expf(-gf));
            float gv = tanhf(gg);
            float og = 1.f / (1.f + __expf(-go));
            c[m] = fg * c[m] + ig * gv;
            float hv = og * tanhf(c[m]);
            __bf16 hb = (__bf16)hv;
            int u = slice * UPW + jj;
            hnext[(size_t)pb * H_ + u] = hb;
            hist[((size_t)tx * B_ + pb) * (2 * H_) + dir * H_ + u] = hb;
        }

        // grid-wide lockstep barrier (32 resident WGs)
        __threadfence();
        __syncthreads();
        if (tid == 0) {
            atomicAdd(ctr, 1u);
            const unsigned target = (unsigned)NWG * (unsigned)(t + 1);
            while (__hip_atomic_load(ctr, __ATOMIC_ACQUIRE, __HIP_MEMORY_SCOPE_AGENT) < target)
                __builtin_amdgcn_s_sleep(2);
        }
        __syncthreads();
    }
}

// ---------------- attention scores: tanh(h @ attW^T) @ v ----------------

__global__ void __launch_bounds__(256) attn_scores(
    const __bf16* __restrict__ hist, const __bf16* __restrict__ attWbf,
    const float* __restrict__ attv, float* __restrict__ scoresT)
{
    const int lane = threadIdx.x & 31, wv = threadIdx.x >> 5;
    const int rt = blockIdx.x * 8 + wv;          // row tile over 32768 (t*64+b) rows
    const int l15 = lane & 15, lh = lane >> 4;
    const __bf16* arow = hist + (size_t)(rt * 16 + l15) * 1024;

    v8f zero = {};
    v8f acc[4];
#pragma unroll
    for (int j = 0; j < 4; ++j) acc[j] = zero;

    for (int kt = 0; kt < 32; ++kt) {
        v8bf a0 = *(const v8bf*)(arow + kt * 32 + 16 * lh);
        v8bf a1 = *(const v8bf*)(arow + kt * 32 + 16 * lh + 8);
        v16bf a = __builtin_shufflevector(a0, a1, 0,1,2,3,4,5,6,7,8,9,10,11,12,13,14,15);
#pragma unroll
        for (int nt = 0; nt < 4; ++nt) {
            v16bf bt = *(const v16bf*)(attWbf + (size_t)(nt * 16 + l15) * 1024 + kt * 32 + 16 * lh);
            acc[nt] = __builtin_amdgcn_wmma_f32_16x16x32_bf16(
                false, a, false, bt, (short)0, acc[nt], false, false);
        }
    }

    float vq[4];
#pragma unroll
    for (int nt = 0; nt < 4; ++nt) vq[nt] = attv[nt * 16 + l15];

#pragma unroll
    for (int r = 0; r < 8; ++r) {
        float s = 0.f;
#pragma unroll
        for (int nt = 0; nt < 4; ++nt) s += tanhf(acc[nt][r]) * vq[nt];
        // reduce over the 16 lanes of this half (cols); rows differ per half
        s += __shfl_xor(s, 1, 32);
        s += __shfl_xor(s, 2, 32);
        s += __shfl_xor(s, 4, 32);
        s += __shfl_xor(s, 8, 32);
        if (l15 == 0) {
            int row = rt * 16 + r + 8 * lh;      // row = t*64 + b
            scoresT[(row & 63) * S_ + (row >> 6)] = s;
        }
    }
}

// ---------------- softmax over time + weighted pooling ----------------

__global__ void __launch_bounds__(256) attn_pool(
    const __bf16* __restrict__ hist, const float* __restrict__ scoresT,
    float* __restrict__ out)
{
    __shared__ float sc[S_];
    const int b = blockIdx.x;
    for (int t = threadIdx.x; t < S_; t += 256) sc[t] = scoresT[b * S_ + t];
    __syncthreads();
    float mx = -1e30f;
    for (int t = 0; t < S_; ++t) mx = fmaxf(mx, sc[t]);
    __syncthreads();
    for (int t = threadIdx.x; t < S_; t += 256) sc[t] = __expf(sc[t] - mx);
    __syncthreads();
    float sum = 0.f;
    for (int t = 0; t < S_; ++t) sum += sc[t];
    const float inv = 1.f / sum;
    for (int j = threadIdx.x; j < 2 * H_; j += 256) {
        float acc = 0.f;
        for (int t = 0; t < S_; ++t)
            acc += sc[t] * (float)hist[((size_t)t * B_ + b) * (2 * H_) + j];
        out[(size_t)b * (2 * H_) + j] = acc * inv;
    }
}

// ---------------- host launcher ----------------

extern "C" void kernel_launch(void* const* d_in, const int* in_sizes, int n_in,
                              void* d_out, int out_size, void* d_ws, size_t ws_size,
                              hipStream_t stream)
{
    (void)in_sizes; (void)n_in; (void)out_size; (void)ws_size;
    const float* x    = (const float*)d_in[0];
    const float* fWih = (const float*)d_in[1];
    const float* fWhh = (const float*)d_in[2];
    const float* fbih = (const float*)d_in[3];
    const float* fbhh = (const float*)d_in[4];
    const float* bWih = (const float*)d_in[5];
    const float* bWhh = (const float*)d_in[6];
    const float* bbih = (const float*)d_in[7];
    const float* bbhh = (const float*)d_in[8];
    const float* attW = (const float*)d_in[9];
    const float* attv = (const float*)d_in[10];
    float* out = (float*)d_out;

    char* base = (char*)d_ws;
    size_t off = 0;
    auto carve = [&](size_t bytes) {
        char* p = base + off;
        off = (off + bytes + 255) & ~(size_t)255;
        return p;
    };
    __bf16*  xbf     = (__bf16*)carve((size_t)S_ * B_ * D_ * 2);          // 16 MB
    __bf16*  WtP     = (__bf16*)carve((size_t)2 * G_ * KTOT * 2);         // 6 MB
    float*   biasP   = (float*) carve((size_t)2 * G_ * 4);
    __bf16*  hbuf    = (__bf16*)carve((size_t)2 * 2 * B_ * H_ * 2);       // double buffered h
    __bf16*  hist    = (__bf16*)carve((size_t)S_ * B_ * 2 * H_ * 2);      // 64 MB
    __bf16*  attWb   = (__bf16*)carve((size_t)64 * 1024 * 2);
    float*   scoresT = (float*) carve((size_t)B_ * S_ * 4);
    unsigned* ctr    = (unsigned*)carve(256);

    hipMemsetAsync(hbuf, 0, (size_t)2 * 2 * B_ * H_ * 2, stream);  // h0 = 0
    hipMemsetAsync(ctr, 0, 256, stream);                           // grid barrier

    pack_weights<<<dim3(4096), dim3(256), 0, stream>>>(fWih, fWhh, fbih, fbhh,
                                                       bWih, bWhh, bbih, bbhh, WtP, biasP);
    pack_x<<<dim3(32768), dim3(256), 0, stream>>>(x, xbf);
    pack_attw<<<dim3(256), dim3(256), 0, stream>>>(attW, attWb);

    hipFuncSetAttribute((const void*)lstm_persistent,
                        hipFuncAttributeMaxDynamicSharedMemorySize, LDS_BYTES);
    lstm_persistent<<<dim3(NWG), dim3(256), LDS_BYTES, stream>>>(WtP, biasP, xbf, hbuf, hist, ctr);

    attn_scores<<<dim3(256), dim3(256), 0, stream>>>(hist, attWb, attv, scoresT);
    attn_pool<<<dim3(64), dim3(256), 0, stream>>>(hist, scoresT, out);
}
